// Encode_78572131713673
// MI455X (gfx1250) — compile-verified, matrix operand order checked by reading, not run on hardware
//
#include <hip/hip_runtime.h>
#include <hip/hip_bf16.h>
#include <hip/hip_fp16.h>

typedef float      v2f  __attribute__((ext_vector_type(2)));
typedef float      v8f  __attribute__((ext_vector_type(8)));
typedef _Float16   v8h  __attribute__((ext_vector_type(8)));
typedef _Float16   v16h __attribute__((ext_vector_type(16)));

#define B_    32
#define T_    256
#define CD    100
#define BD    100
#define TD    20
#define DIN   420
#define LIN   200
#define H_    200
#define NU    208      // units padded to multiple of 16
#define NG    832      // 4 * NU (permuted gate rows)
#define KH    224      // recurrence K (hidden) padded to multiple of 32
#define HS    232      // LDS row stride for h (halves) - bank-friendly
#define XS    212      // LDS row stride for x tile (floats) - bank-friendly
#define AS    426      // LDS row stride for staged A rows (floats) - bank-friendly
#define MTOK  (B_ * T_)  // 8192 tokens

// Branch-free activations on hardware v_exp_f32 + single v_rcp_f32 (no IEEE
// div_scale/div_fixup sequence). rcp(inf)=0 preserves saturation limits:
// tanh -> +/-1, sigmoid -> 0/1.
__device__ __forceinline__ float fast_tanh(float x) {
    float e = __expf(2.f * x);
    return 1.f - 2.f * __builtin_amdgcn_rcpf(e + 1.f);
}
__device__ __forceinline__ float fast_sigmoid(float x) {
    return __builtin_amdgcn_rcpf(1.f + __expf(-x));
}

// ---------------------------------------------------------------------------
// Kernel 1: permute + pad weights into gate-interleaved layout.
//   row r' = ublk*64 + gate*16 + uin  (unit u = ublk*16+uin, padded u>=200 -> 0)
//   Wihp: [832,200] f32,  bp: [832] f32,  Whhp: [832,224] f16 (K padded w/ 0)
// ---------------------------------------------------------------------------
__global__ void prep_weights(const float* __restrict__ Wih_l, const float* __restrict__ Whh_l,
                             const float* __restrict__ b_l,
                             const float* __restrict__ Wih_r, const float* __restrict__ Whh_r,
                             const float* __restrict__ b_r,
                             float* __restrict__ Wihp_l, float* __restrict__ Wihp_r,
                             float* __restrict__ bp_l,   float* __restrict__ bp_r,
                             _Float16* __restrict__ Whhp_l, _Float16* __restrict__ Whhp_r)
{
    int idx = blockIdx.x * blockDim.x + threadIdx.x;
    if (idx >= NG * KH) return;
    int r = idx / KH, k = idx % KH;
    int ublk = r >> 6, rem = r & 63, gate = rem >> 4, uin = rem & 15;
    int u = ublk * 16 + uin;
    bool valid = (u < H_);
    int ro = gate * H_ + u;   // original row in [4H, *] matrices

    _Float16 wl = (_Float16)0.f, wr = (_Float16)0.f;
    if (valid && k < H_) {
        wl = (_Float16)Whh_l[ro * H_ + k];
        wr = (_Float16)Whh_r[ro * H_ + k];
    }
    Whhp_l[r * KH + k] = wl;
    Whhp_r[r * KH + k] = wr;

    if (k < LIN) {
        float il = 0.f, ir = 0.f;
        if (valid) { il = Wih_l[ro * LIN + k]; ir = Wih_r[ro * LIN + k]; }
        Wihp_l[r * LIN + k] = il;
        Wihp_r[r * LIN + k] = ir;
    }
    if (k == 0) {
        bp_l[r] = valid ? b_l[ro] : 0.f;
        bp_r[r] = valid ? b_r[ro] : 0.f;
    }
}

// ---------------------------------------------------------------------------
// Kernel 2: fused embed-gather + FC(420->200)+tanh + gx = x @ Wihp^T + bp.
// One block = (side, 16-token M-tile). FP32 WMMA 16x16x4 throughout.
//   phase 0: cooperative coalesced gather of 16 x 420 embedding rows -> LDS
//   phase 1: x = tanh(A @ fc_w^T + fc_b) -> LDS (branch-free inner loop)
//   phase 2: gx = x @ Wihp^T + bp        -> global (L2-resident for kernel 3)
// Token ordering m = t*32 + b so the recurrence reads 32 consecutive rows.
// ---------------------------------------------------------------------------
__global__ __launch_bounds__(256) void encode_gemm(
    const int* __restrict__ chars,   const int* __restrict__ lb,  const int* __restrict__ rb,
    const int* __restrict__ echars,  const int* __restrict__ elb, const int* __restrict__ erb,
    const int* __restrict__ ctype,
    const float* __restrict__ E_char,  const float* __restrict__ E_extchar,
    const float* __restrict__ E_bichar,const float* __restrict__ E_extbichar,
    const float* __restrict__ E_ctype,
    const float* __restrict__ fc_w, const float* __restrict__ fc_b,
    const float* __restrict__ Wihp_l, const float* __restrict__ bp_l,
    const float* __restrict__ Wihp_r, const float* __restrict__ bp_r,
    float* __restrict__ gxl, float* __restrict__ gxr)
{
    __shared__ __align__(16) float at_[16 * AS];  // staged A rows (16 x 420)
    __shared__ __align__(16) float xt[16 * XS];   // tanh(FC) tile   (16 x 208)

    int side = blockIdx.x >> 9;      // 0 = left, 1 = right
    int m0   = (blockIdx.x & 511) * 16;
    int lane = threadIdx.x & 31;
    int wave = threadIdx.x >> 5;     // 8 waves
    int lr   = lane & 15;
    int kk   = (lane < 16) ? 0 : 2;  // f32 A/B frag: half-wave holds K {0,1} vs {2,3}

    const int* bi  = side ? rb  : lb;
    const int* ebi = side ? erb : elb;

    // ---- Phase 0: gather 16 tokens x 420 features into LDS (coalesced) ----
    for (int idx = threadIdx.x; idx < 16 * DIN; idx += 256) {
        int row = idx / DIN, col = idx - row * DIN;
        int mm = m0 + row;
        int tix = (mm & 31) * T_ + (mm >> 5);   // chars[b, t]
        float v;
        if      (col < CD)             v = E_char     [(size_t)chars [tix] * CD + col];
        else if (col < 2*CD)           v = E_extchar  [(size_t)echars[tix] * CD + (col - CD)];
        else if (col < 2*CD + BD)      v = E_bichar   [(size_t)bi    [tix] * BD + (col - 2*CD)];
        else if (col < 2*CD + 2*BD)    v = E_extbichar[(size_t)ebi   [tix] * BD + (col - 2*CD - BD)];
        else                           v = E_ctype    [(size_t)ctype [tix] * TD + (col - 2*CD - 2*BD)];
        at_[row * AS + col] = v;
    }
    __syncthreads();

    // ---- Phase 1: x = tanh(A @ fc_w^T + fc_b) into LDS x tile ----
    // fc_w row index clamped for n in [200,208): those columns land in LDS
    // padding that phase 2 never reads (its K-loop stops at 200).
    for (int nt = wave; nt < 13; nt += 8) {
        int n  = nt * 16 + lr;
        int nc = (n < LIN) ? n : (LIN - 1);     // clamp, branch-free hot loop
        const float* fwp = fc_w + (size_t)nc * DIN;
        v8f acc = {};
#pragma unroll 5
        for (int k0 = 0; k0 < DIN; k0 += 4) {
            int kb = k0 + kk;
            v2f a;  a.x  = at_[lr * AS + kb];  a.y  = at_[lr * AS + kb + 1];
            v2f bf; bf.x = fwp[kb];            bf.y = fwp[kb + 1];
            acc = __builtin_amdgcn_wmma_f32_16x16x4_f32(false, a, false, bf,
                                                        (short)0, acc, false, false);
        }
        float bias = fc_b[nc];
#pragma unroll
        for (int r = 0; r < 8; ++r) {
            int mr = (lane < 16) ? r : (8 + r);
            xt[mr * XS + n] = fast_tanh(acc[r] + bias);
        }
    }
    __syncthreads();

    // ---- Phase 2: gx[m, 0:832] = x @ Wihp^T + bp ----
    const float* Wp = side ? Wihp_r : Wihp_l;
    const float* bp = side ? bp_r   : bp_l;
    float*       gx = side ? gxr    : gxl;
    for (int nt = wave; nt < NG / 16; nt += 8) {
        int n = nt * 16 + lr;
        const float* wpp = Wp + (size_t)n * LIN;
        v8f acc = {};
#pragma unroll 5
        for (int k0 = 0; k0 < LIN; k0 += 4) {
            int kb = k0 + kk;
            v2f a;  a.x  = xt[lr * XS + kb];  a.y  = xt[lr * XS + kb + 1];
            v2f bf; bf.x = wpp[kb];           bf.y = wpp[kb + 1];
            acc = __builtin_amdgcn_wmma_f32_16x16x4_f32(false, a, false, bf,
                                                        (short)0, acc, false, false);
        }
        float bias = bp[n];
#pragma unroll
        for (int r = 0; r < 8; ++r) {
            int mr = (lane < 16) ? r : (8 + r);
            gx[(size_t)(m0 + mr) * NG + n] = acc[r] + bias;
        }
    }
}

// ---------------------------------------------------------------------------
// Kernel 3: persistent LSTM recurrence. grid = 2 blocks (dir 0 fwd / 1 rev),
// 832 threads = 26 waves; wave (mt, ublk) owns a 16-batch x 16-unit chunk:
// its 4 WMMA tiles are exactly gates i/f/g/o -> all gate math in registers,
// fully branch-free per step (v_exp + v_rcp only, no div sequences).
// h double-buffered f16 in LDS; c lives in accumulator-layout VGPRs.
// FP16 WMMA 16x16x32, K = 224 -> 7 MACs per tile, 28 per wave per step.
// ---------------------------------------------------------------------------
__global__ __launch_bounds__(832) void lstm_rec(
    const float* __restrict__ gxl, const float* __restrict__ gxr,
    const _Float16* __restrict__ Whhp_l, const _Float16* __restrict__ Whhp_r,
    float* __restrict__ out)
{
    __shared__ __align__(16) _Float16 hbuf[2][32 * HS];

    int dir = blockIdx.x;
    const float*    gx = dir ? gxr    : gxl;
    const _Float16* Wp = dir ? Whhp_r : Whhp_l;

    for (int i = threadIdx.x; i < 2 * 32 * HS; i += blockDim.x)
        (&hbuf[0][0])[i] = (_Float16)0.f;
    __syncthreads();

    int lane = threadIdx.x & 31;
    int wave = threadIdx.x >> 5;     // 0..25
    int mt   = wave & 1;             // batch tile (rows 0-15 / 16-31)
    int ublk = wave >> 1;            // unit block 0..12
    int m0   = mt * 16;
    int lr   = lane & 15;
    int koff = (lane < 16) ? 0 : 8;  // f16 A/B frag K offset per half-wave
    int u    = ublk * 16 + lr;       // this lane's unit column

    v8f cst = {};   // cell state, accumulator layout (16x16 chunk / 32 lanes)

    for (int s = 0; s < T_; ++s) {
        int t   = dir ? (T_ - 1 - s) : s;
        int cur = s & 1, nxt = cur ^ 1;
        const _Float16* hb = hbuf[cur];

        v8f acc[4] = {};
#pragma unroll
        for (int k0 = 0; k0 < KH; k0 += 32) {
            v8h alo = *(const v8h*)&hb[(m0 + lr) * HS + k0 + koff];
            v8h ahi = *(const v8h*)&hb[(m0 + lr) * HS + k0 + 16 + koff];
            v16h a  = __builtin_shufflevector(alo, ahi, 0,1,2,3,4,5,6,7,8,9,10,11,12,13,14,15);
#pragma unroll
            for (int g = 0; g < 4; ++g) {
                int n = ublk * 64 + g * 16 + lr;
                v8h blo = *(const v8h*)&Wp[(size_t)n * KH + k0 + koff];
                v8h bhi = *(const v8h*)&Wp[(size_t)n * KH + k0 + 16 + koff];
                v16h bb = __builtin_shufflevector(blo, bhi, 0,1,2,3,4,5,6,7,8,9,10,11,12,13,14,15);
                acc[g] = __builtin_amdgcn_wmma_f32_16x16x32_f16(false, a, false, bb,
                                                                (short)0, acc[g], false, false);
            }
        }

        // add precomputed input-gate contribution gx[t]
#pragma unroll
        for (int g = 0; g < 4; ++g) {
            int n = ublk * 64 + g * 16 + lr;
            const float* gp = gx + (size_t)t * 32 * NG + n;
#pragma unroll
            for (int r = 0; r < 8; ++r) {
                int mr = m0 + ((lane < 16) ? r : 8 + r);
                acc[g][r] += gp[(size_t)mr * NG];
            }
        }

        // gates: i,f,g,o are acc[0..3], element-aligned across tiles
        v8f hnew = {};
#pragma unroll
        for (int r = 0; r < 8; ++r) {
            float ig = fast_sigmoid(acc[0][r]);
            float fg = fast_sigmoid(acc[1][r]);
            float gg = fast_tanh(acc[2][r]);
            float og = fast_sigmoid(acc[3][r]);
            float cv = fg * cst[r] + ig * gg;
            cst[r]   = cv;
            hnew[r]  = og * fast_tanh(cv);
        }

        // write h into the other LDS buffer + the valid slice to global out
        _Float16* hn = hbuf[nxt];
#pragma unroll
        for (int r = 0; r < 8; ++r) {
            int mr = m0 + ((lane < 16) ? r : 8 + r);
            hn[mr * HS + u] = (_Float16)hnew[r];
            if (u < H_)
                out[((size_t)mr * T_ + t) * (2 * H_) + dir * H_ + u] = hnew[r];
        }
        __syncthreads();   // ping-pong: one barrier per step
    }
}

// ---------------------------------------------------------------------------
extern "C" void kernel_launch(void* const* d_in, const int* in_sizes, int n_in,
                              void* d_out, int out_size, void* d_ws, size_t ws_size,
                              hipStream_t stream)
{
    (void)in_sizes; (void)n_in; (void)out_size; (void)ws_size;

    const int* chars  = (const int*)d_in[0];
    const int* lb     = (const int*)d_in[1];
    const int* rb     = (const int*)d_in[2];
    const int* echars = (const int*)d_in[3];
    const int* elb    = (const int*)d_in[4];
    const int* erb    = (const int*)d_in[5];
    const int* ctype  = (const int*)d_in[6];
    const float* E_char      = (const float*)d_in[7];
    const float* E_extchar   = (const float*)d_in[8];
    const float* E_bichar    = (const float*)d_in[9];
    const float* E_extbichar = (const float*)d_in[10];
    const float* E_ctype     = (const float*)d_in[11];
    const float* fc_w  = (const float*)d_in[12];
    const float* fc_b  = (const float*)d_in[13];
    const float* Wih_l = (const float*)d_in[14];
    const float* Whh_l = (const float*)d_in[15];
    const float* b_l   = (const float*)d_in[16];
    const float* Wih_r = (const float*)d_in[17];
    const float* Whh_r = (const float*)d_in[18];
    const float* b_r   = (const float*)d_in[19];

    char*  ws  = (char*)d_ws;
    size_t off = 0;
    auto alloc = [&](size_t bytes) -> void* {
        void* p = ws + off;
        off += (bytes + 255) & ~(size_t)255;
        return p;
    };
    float*    gxl    = (float*)   alloc((size_t)MTOK * NG * sizeof(float));   // ~27 MB
    float*    gxr    = (float*)   alloc((size_t)MTOK * NG * sizeof(float));   // ~27 MB
    float*    Wihp_l = (float*)   alloc((size_t)NG * LIN * sizeof(float));
    float*    Wihp_r = (float*)   alloc((size_t)NG * LIN * sizeof(float));
    float*    bp_l   = (float*)   alloc((size_t)NG * sizeof(float));
    float*    bp_r   = (float*)   alloc((size_t)NG * sizeof(float));
    _Float16* Whhp_l = (_Float16*)alloc((size_t)NG * KH * sizeof(_Float16));
    _Float16* Whhp_r = (_Float16*)alloc((size_t)NG * KH * sizeof(_Float16));

    prep_weights<<<(NG * KH + 255) / 256, 256, 0, stream>>>(
        Wih_l, Whh_l, b_l, Wih_r, Whh_r, b_r,
        Wihp_l, Wihp_r, bp_l, bp_r, Whhp_l, Whhp_r);

    encode_gemm<<<1024, 256, 0, stream>>>(
        chars, lb, rb, echars, elb, erb, ctype,
        E_char, E_extchar, E_bichar, E_extbichar, E_ctype,
        fc_w, fc_b, Wihp_l, bp_l, Wihp_r, bp_r, gxl, gxr);

    lstm_rec<<<2, 832, 0, stream>>>(gxl, gxr, Whhp_l, Whhp_r, (float*)d_out);
}
